// GCN_67164698575457
// MI455X (gfx1250) — compile-verified, matrix operand order checked by reading, not run on hardware
//
#include <hip/hip_runtime.h>

#define N_NODES 100000
#define N_EDGES 1600000
#define IN_DIM  165
#define HIDDEN  128
#define OUT_DIM 2
#define M_TILES (N_NODES / 16)   // 6250 (divides exactly)
#define K_CHUNKS 6               // ceil(165/32) -> K padded to 192

typedef __attribute__((ext_vector_type(16))) _Float16 v16h;
typedef __attribute__((ext_vector_type(8)))  float    v8f;

// ---------------------------------------------------------------------------
// Degree / normalization:  deg[v] = 1 (self loop) + indegree(v);  dinv = rsqrt
// ---------------------------------------------------------------------------
__global__ void gcn_deg_init(float* deg) {
    int i = blockIdx.x * blockDim.x + threadIdx.x;
    if (i < N_NODES) deg[i] = 1.0f;
}

__global__ void gcn_deg_count(const int* __restrict__ dst, float* deg) {
    int e = blockIdx.x * blockDim.x + threadIdx.x;
    if (e < N_EDGES) atomicAdd(&deg[dst[e]], 1.0f);
}

__global__ void gcn_deg_rsqrt(float* deg) {
    int i = blockIdx.x * blockDim.x + threadIdx.x;
    if (i < N_NODES) {
        float d = deg[i];
        deg[i] = (d > 0.0f) ? rsqrtf(d) : 0.0f;
    }
}

// ---------------------------------------------------------------------------
// GEMM1: h = x @ W1   [100000,165] x [165,128], f16 inputs, f32 accumulate.
// Block = 256 threads = 8 waves; wave w owns N-tile w; grid-stride over M-tiles.
// Each wave preloads its 6 K-chunk B fragments of W1 into registers.
// ---------------------------------------------------------------------------
__global__ void gcn_gemm1_wmma(const float* __restrict__ x,
                               const float* __restrict__ W1,
                               float* __restrict__ h) {
    const int lane  = threadIdx.x & 31;
    const int nTile = threadIdx.x >> 5;           // 0..7
    const int col   = nTile * 16 + (lane & 15);   // output column (N)
    const int hi    = (lane >= 16) ? 8 : 0;       // lane-half K offset
    const int row   = lane & 15;                  // A-matrix row within tile

    // Preload B fragments (W1 is [K=165 rows, N=128 cols], row-major).
    // 16-bit B layout mirrors A: lane half selects K+8; elems 8..15 are K+16.
    v16h bfrag[K_CHUNKS];
#pragma unroll
    for (int c = 0; c < K_CHUNKS; ++c) {
        const int k0 = c * 32;
        v16h b;
#pragma unroll
        for (int i = 0; i < 16; ++i) {
            const int k = k0 + (i & 7) + ((i >= 8) ? 16 : 0) + hi;
            float v = (k < IN_DIM) ? W1[k * HIDDEN + col] : 0.0f;
            b[i] = (_Float16)v;
        }
        bfrag[c] = b;
    }

    for (int mt = blockIdx.x; mt < M_TILES; mt += gridDim.x) {
        const float* xr = x + (size_t)(mt * 16 + row) * IN_DIM;
        v8f acc = {};
#pragma unroll
        for (int c = 0; c < K_CHUNKS; ++c) {
            const int k0 = c * 32;
            v16h a;
#pragma unroll
            for (int i = 0; i < 16; ++i) {
                const int k = k0 + (i & 7) + ((i >= 8) ? 16 : 0) + hi;
                float v = (k < IN_DIM) ? xr[k] : 0.0f;
                a[i] = (_Float16)v;
            }
            // D = A x B + C   (v_wmma_f32_16x16x32_f16)
            acc = __builtin_amdgcn_wmma_f32_16x16x32_f16(
                false, a, false, bfrag[c], (short)0, acc, false, false);
        }
        // C/D layout: VGPR r -> M = r (+8 for hi lanes); N = lane & 15.
        // One base address per tile; r contributes a constant byte offset
        // (r*HIDDEN*4 = r*512 B) so the backend emits offset-immediate stores.
        float* hp = h + (size_t)(mt * 16 + hi) * HIDDEN + col;
#pragma unroll
        for (int r = 0; r < 8; ++r) {
            hp[(size_t)r * HIDDEN] = acc[r];
        }
    }
}

// ---------------------------------------------------------------------------
// Aggregation layer 1 (128 features).
// init: agg[v] = h[v] * dinv[v]^2   (self-loop term)
// edges: agg[d] += h[s] * dinv[s]*dinv[d]  via 32 lanes/edge, 4 feats/lane.
// ---------------------------------------------------------------------------
__global__ void gcn_agg1_init(const float* __restrict__ h,
                              const float* __restrict__ dinv,
                              float* __restrict__ agg) {
    int idx = blockIdx.x * blockDim.x + threadIdx.x;   // N*128 = 12.8M
    if (idx < N_NODES * HIDDEN) {
        int v = idx >> 7;
        float di = dinv[v];
        agg[idx] = h[idx] * di * di;
    }
}

__global__ void gcn_agg1_edges(const float* __restrict__ h,
                               const int* __restrict__ src,
                               const int* __restrict__ dst,
                               const float* __restrict__ dinv,
                               float* __restrict__ agg) {
    long long t = (long long)blockIdx.x * blockDim.x + threadIdx.x;
    int e    = (int)(t >> 5);
    int lane = (int)(t & 31);
    if (e >= N_EDGES) return;
    int s = src[e], d = dst[e];
    float norm = dinv[s] * dinv[d];
    float4 hv = *(const float4*)(h + (size_t)s * HIDDEN + lane * 4);  // 16B-aligned
    float* od = agg + (size_t)d * HIDDEN + lane * 4;
    atomicAdd(od + 0, hv.x * norm);
    atomicAdd(od + 1, hv.y * norm);
    atomicAdd(od + 2, hv.z * norm);
    atomicAdd(od + 3, hv.w * norm);
}

// ---------------------------------------------------------------------------
// Layer 2 linear with fused bias + ReLU on the input:
//   z[v] = relu(agg1[v] + b1) @ W2    (W2: [128,2])
// ---------------------------------------------------------------------------
__global__ void gcn_layer2_gemv(const float* __restrict__ agg,
                                const float* __restrict__ b1,
                                const float* __restrict__ W2,
                                float* __restrict__ z) {
    int v = blockIdx.x * blockDim.x + threadIdx.x;
    if (v >= N_NODES) return;
    const float4* r4 = (const float4*)(agg + (size_t)v * HIDDEN);
    const float4* b4 = (const float4*)b1;
    const float2* w2 = (const float2*)W2;
    float acc0 = 0.0f, acc1 = 0.0f;
#pragma unroll 8
    for (int q = 0; q < HIDDEN / 4; ++q) {
        float4 rv = r4[q];
        float4 bv = b4[q];
        float tv[4] = { rv.x + bv.x, rv.y + bv.y, rv.z + bv.z, rv.w + bv.w };
#pragma unroll
        for (int j = 0; j < 4; ++j) {
            float tt = fmaxf(tv[j], 0.0f);
            float2 w = w2[q * 4 + j];
            acc0 = fmaf(tt, w.x, acc0);
            acc1 = fmaf(tt, w.y, acc1);
        }
    }
    z[2 * v + 0] = acc0;
    z[2 * v + 1] = acc1;
}

// ---------------------------------------------------------------------------
// Aggregation layer 2 (2 features) directly into d_out.
// ---------------------------------------------------------------------------
__global__ void gcn_out_init(const float* __restrict__ z,
                             const float* __restrict__ dinv,
                             const float* __restrict__ b2,
                             float* __restrict__ out) {
    int v = blockIdx.x * blockDim.x + threadIdx.x;
    if (v >= N_NODES) return;
    float di = dinv[v];
    float n = di * di;
    out[2 * v + 0] = z[2 * v + 0] * n + b2[0];
    out[2 * v + 1] = z[2 * v + 1] * n + b2[1];
}

__global__ void gcn_out_edges(const float* __restrict__ z,
                              const int* __restrict__ src,
                              const int* __restrict__ dst,
                              const float* __restrict__ dinv,
                              float* __restrict__ out) {
    int e = blockIdx.x * blockDim.x + threadIdx.x;
    if (e >= N_EDGES) return;
    int s = src[e], d = dst[e];
    float norm = dinv[s] * dinv[d];
    atomicAdd(&out[2 * d + 0], z[2 * s + 0] * norm);
    atomicAdd(&out[2 * d + 1], z[2 * s + 1] * norm);
}

// ---------------------------------------------------------------------------
extern "C" void kernel_launch(void* const* d_in, const int* in_sizes, int n_in,
                              void* d_out, int out_size, void* d_ws, size_t ws_size,
                              hipStream_t stream) {
    const float* x  = (const float*)d_in[0];            // [100000,165]
    const int*   ei = (const int*)d_in[1];              // [2,1600000]
    const float* W1 = (const float*)d_in[2];            // [165,128]
    const float* b1 = (const float*)d_in[3];            // [128]
    const float* W2 = (const float*)d_in[4];            // [128,2]
    const float* b2 = (const float*)d_in[5];            // [2]
    float* out = (float*)d_out;                         // [100000,2]

    const int* src = ei;
    const int* dst = ei + N_EDGES;

    // Workspace partition (all offsets 256B-aligned; d_ws is 256B-aligned).
    char* ws = (char*)d_ws;
    size_t off = 0;
    float* h    = (float*)(ws + off); off += (size_t)N_NODES * HIDDEN * sizeof(float);
    float* agg1 = (float*)(ws + off); off += (size_t)N_NODES * HIDDEN * sizeof(float);
    float* z    = (float*)(ws + off); off += (size_t)N_NODES * 2 * sizeof(float) + 512;
    float* dinv = (float*)(ws + off); off += (size_t)N_NODES * sizeof(float);
    (void)ws_size; (void)in_sizes; (void)n_in; (void)out_size;

    const int T = 256;
    const int gN   = (N_NODES + T - 1) / T;
    const int gE   = (N_EDGES + T - 1) / T;
    const int gNF  = (N_NODES * HIDDEN + T - 1) / T;
    const int gE32 = (int)(((long long)N_EDGES * 32 + T - 1) / T);

    // 1) symmetric-normalization coefficients
    gcn_deg_init <<<gN, T, 0, stream>>>(dinv);
    gcn_deg_count<<<gE, T, 0, stream>>>(dst, dinv);
    gcn_deg_rsqrt<<<gN, T, 0, stream>>>(dinv);

    // 2) h = x @ W1  (WMMA)
    gcn_gemm1_wmma<<<1024, 256, 0, stream>>>(x, W1, h);

    // 3) layer-1 aggregation (self-loop init + edge scatter-add)
    gcn_agg1_init <<<gNF,  T, 0, stream>>>(h, dinv, agg1);
    gcn_agg1_edges<<<gE32, T, 0, stream>>>(h, src, dst, dinv, agg1);

    // 4) z = relu(agg1 + b1) @ W2
    gcn_layer2_gemv<<<gN, T, 0, stream>>>(agg1, b1, W2, z);

    // 5) layer-2 aggregation straight into d_out (+ bias)
    gcn_out_init <<<gN, T, 0, stream>>>(z, dinv, b2, out);
    gcn_out_edges<<<gE, T, 0, stream>>>(z, src, dst, dinv, out);
}
